// MultiHeadSelfAttentionBlock_28020366639418
// MI455X (gfx1250) — matrix operand order repaired and checked by reading, not compiled, not run on hardware
//
#include <hip/hip_runtime.h>
#include <hip/hip_bf16.h>
#include <math.h>

// ---------------------------------------------------------------------------
// MI455X (gfx1250) implementation of the MQA block.
// All contractions run on v_wmma_f32_16x16x32_bf16 (bf16 in, f32 accumulate).
// Softmax is over the HEAD axis (8 values) -> purely local per (m,n), so a
// single pass over m with per-tile softmax in LDS suffices (no running max).
// K/V tiles are staged into double-buffered LDS with the gfx1250 async
// global->LDS path (ASYNCcnt) so the 8 waves of a workgroup share one copy.
// ---------------------------------------------------------------------------

typedef __bf16 v16bf __attribute__((ext_vector_type(16)));
typedef __bf16 v8bf  __attribute__((ext_vector_type(8)));
typedef float  v8f   __attribute__((ext_vector_type(8)));
typedef int    v4i   __attribute__((__vector_size__(16)));

#define B_  8
#define C_  256
#define N_  1024   // H*W
#define NH_ 8
#define KD_ 64
#define VD_ 64

#if __has_builtin(__builtin_amdgcn_global_load_async_to_lds_b128)
#define HAVE_ASYNC_LDS 1
#else
#define HAVE_ASYNC_LDS 0
#endif

// Stage 16 bytes (8 bf16) from global to LDS, async if available.
// Builtin signature (from hipcc diagnostic): param0 = v4i in AS1 (__device__),
// param1 = v4i in AS3 (__shared__), then imm offset, imm cpol.
__device__ __forceinline__ void stage16(const __bf16* g, __bf16* l) {
#if HAVE_ASYNC_LDS
  __builtin_amdgcn_global_load_async_to_lds_b128(
      (__attribute__((address_space(1))) v4i*)(__bf16*)g,
      (__attribute__((address_space(3))) v4i*)l, 0, 0);
#else
  *(v8bf*)l = *(const v8bf*)g;
#endif
}

__device__ __forceinline__ void async_wait_le2() {
#if HAVE_ASYNC_LDS
#if __has_builtin(__builtin_amdgcn_s_wait_asynccnt)
  __builtin_amdgcn_s_wait_asynccnt(2);
#else
  asm volatile("s_wait_asynccnt 2" ::: "memory");
#endif
#endif
}

__device__ __forceinline__ void async_wait_le0() {
#if HAVE_ASYNC_LDS
#if __has_builtin(__builtin_amdgcn_s_wait_asynccnt)
  __builtin_amdgcn_s_wait_asynccnt(0);
#else
  asm volatile("s_wait_asynccnt 0" ::: "memory");
#endif
#endif
}

union V16U { v16bf v; v8bf h[2]; };

// A-operand (16x32, 16-bit): lane = row (lane&15); the lane's 16 elements
// cover K = base..base+15 contiguous with base = (lane>=16 ? 16 : 0).
// Caller passes rowptr already offset by the per-lane half (halfA).
__device__ __forceinline__ v16bf loadA(const __bf16* rowptr) {
  V16U u;
  u.h[0] = *(const v8bf*)(rowptr);
  u.h[1] = *(const v8bf*)(rowptr + 8);
  return u.v;
}

// B-operand (32x16, 16-bit): lane = column (lane&15); the lane's 16 elements
// are K contiguous, lanes 0-15 cover K 0..15, lanes 16-31 cover K 16..31.
// Caller passes colptr already offset by halfB (0 or 16 elements).
__device__ __forceinline__ v16bf loadB(const __bf16* colptr) {
  return *(const v16bf*)colptr;
}

__device__ __forceinline__ v8f wmma_bf16(v16bf a, v16bf b, v8f c) {
  return __builtin_amdgcn_wmma_f32_16x16x32_bf16(
      /*neg_a=*/false, a, /*neg_b=*/false, b,
      /*c_mod=*/(short)0, c, /*reuse_a=*/false, /*reuse_b=*/false);
}

// ---------------------------------------------------------------------------
// Prep 1: fold BatchNorm into x, write bf16 xnT[b][n][d] (d contiguous).
// ---------------------------------------------------------------------------
__global__ __launch_bounds__(256) void prep_xnT_kernel(
    const float* __restrict__ x, const float* __restrict__ g,
    const float* __restrict__ be, const float* __restrict__ mu,
    const float* __restrict__ var, __bf16* __restrict__ xnT) {
  size_t id = (size_t)blockIdx.x * 256 + threadIdx.x;
  if (id >= (size_t)B_ * N_ * C_) return;
  int d = (int)(id % C_);
  size_t bn = id / C_;
  int n = (int)(bn % N_);
  int b = (int)(bn / N_);
  float sc = g[d] * rsqrtf(var[d] + 1e-3f);
  float sh = be[d] - mu[d] * sc;
  float xv = x[((size_t)b * C_ + d) * N_ + n];
  xnT[id] = (__bf16)(xv * sc + sh);
}

// ---------------------------------------------------------------------------
// Prep 2: bf16 weights in A-operand (contraction-dim contiguous) layouts.
//   wqT[(h*64+k)][d] = Wq[h][d][k]
//   wkT[k][d]        = Wk[d][k]
//   wvT[v][d]        = Wv[d][v]
//   wo [d][v*8+h]    = Wo[d][v][h]  (contiguous copy-convert)
// ---------------------------------------------------------------------------
__global__ __launch_bounds__(256) void prep_w_kernel(
    const float* __restrict__ Wq, const float* __restrict__ Wk,
    const float* __restrict__ Wv, const float* __restrict__ Wo,
    __bf16* __restrict__ wqT, __bf16* __restrict__ wkT,
    __bf16* __restrict__ wvT, __bf16* __restrict__ wo_bf) {
  int i = blockIdx.x * 256 + threadIdx.x;
  if (i < 131072) {                       // 512*256
    int r = i >> 8, d = i & 255;
    int h = r >> 6, kk = r & 63;
    wqT[i] = (__bf16)Wq[((size_t)h * C_ + d) * KD_ + kk];
  } else if (i < 147456) {                // + 64*256
    int j = i - 131072; int kk = j >> 8, d = j & 255;
    wkT[j] = (__bf16)Wk[(size_t)d * KD_ + kk];
  } else if (i < 163840) {                // + 64*256
    int j = i - 147456; int vv = j >> 8, d = j & 255;
    wvT[j] = (__bf16)Wv[(size_t)d * VD_ + vv];
  } else if (i < 294912) {                // + 256*512
    int j = i - 163840;
    wo_bf[j] = (__bf16)Wo[j];
  }
}

// ---------------------------------------------------------------------------
// QKV: per (b, n-tile of 16), GEMM [640 x 256] x [256 x 16] on WMMA.
// The shared 16x256 Xn B-tile is staged once into LDS (async) and reused by
// all 8 waves x 5 row-tiles.
// Rows 0..511 -> q (stored [b][h][n][k], k contiguous = B-operand for attn)
// Rows 512..575 -> k (stored [b][m][k], k contiguous = A-operand for logits)
// Rows 576..639 -> v (stored [b][v][m], m contiguous = A-operand for o-GEMM)
// ---------------------------------------------------------------------------
__global__ __launch_bounds__(256) void qkv_kernel(
    const __bf16* __restrict__ xnT, const __bf16* __restrict__ wqT,
    const __bf16* __restrict__ wkT, const __bf16* __restrict__ wvT,
    __bf16* __restrict__ q_ws, __bf16* __restrict__ k_ws,
    __bf16* __restrict__ v_ws) {
  __shared__ __bf16 x_lds[16][256];   // 8 KB

  const int b    = blockIdx.x >> 6;
  const int n0   = (blockIdx.x & 63) << 4;
  const int wave = threadIdx.x >> 5;
  const int lane = threadIdx.x & 31;
  const int lrow = lane & 15;
  const int halfA = (lane >> 4) << 4;   // 0 or 16 elements
  const int halfB = (lane >> 4) << 4;   // 0 or 16 elements
  const int mhi  = (lane >> 4) << 3;    // +8 row offset for hi half-lanes

  // Cooperative stage of the Xn tile: 16 rows x 256 cols = 512 x 16B chunks.
  {
    const int t = (int)threadIdx.x;
#pragma unroll
    for (int p = 0; p < 2; ++p) {
      int c = t + p * 256;
      int r = c >> 5, cc = (c & 31) * 8;
      stage16(xnT + ((size_t)b * N_ + n0 + r) * C_ + cc, &x_lds[r][cc]);
    }
    async_wait_le0();
    __syncthreads();
  }

  const __bf16* xrow = &x_lds[lrow][0];

  for (int i = 0; i < 5; ++i) {
    const int r0 = (wave * 5 + i) * 16;         // 16-row tile, never crosses 512/576
    const __bf16* abase;
    if (r0 < 512)      abase = wqT + (size_t)r0 * C_;
    else if (r0 < 576) abase = wkT + (size_t)(r0 - 512) * C_;
    else               abase = wvT + (size_t)(r0 - 576) * C_;
    const __bf16* arow = abase + (size_t)lrow * C_;

    v8f acc = {};
#pragma unroll
    for (int ks = 0; ks < 8; ++ks) {           // K = C = 256 -> 8 steps of 32
      acc = wmma_bf16(loadA(arow + ks * 32 + halfA),
                      loadB(xrow + ks * 32 + halfB), acc);
    }
#pragma unroll
    for (int j = 0; j < 8; ++j) {
      int r = r0 + j + mhi;
      int n = n0 + lrow;
      __bf16 val = (__bf16)acc[j];
      if (r < 512) {
        int h = r >> 6, kk = r & 63;
        q_ws[(((size_t)(b * NH_ + h) * N_) + n) * KD_ + kk] = val;
      } else if (r < 576) {
        k_ws[((size_t)b * N_ + n) * KD_ + (r - 512)] = val;
      } else {
        v_ws[((size_t)(b * VD_ + (r - 576))) * N_ + n] = val;
      }
    }
  }
}

// ---------------------------------------------------------------------------
// Attention + output projection + LayerScale + residual.
// Workgroup = 8 waves = 8 heads, one (b, 16-wide n-tile).
// K/V tiles double-buffered in LDS via async global->LDS DMA.
// ---------------------------------------------------------------------------
__global__ __launch_bounds__(256) void attn_kernel(
    const __bf16* __restrict__ q_ws, const __bf16* __restrict__ k_ws,
    const __bf16* __restrict__ v_ws, const __bf16* __restrict__ wo_bf,
    const float* __restrict__ x, const float* __restrict__ ls,
    float* __restrict__ out) {
  __shared__ float  logits_lds[32][16][8];   // [m][n][h]  16 KB
  __shared__ __bf16 attn_lds[8][16][32];     // [h][n][m]   8 KB
  __shared__ __bf16 o_lds[16][512];          // [n][v*8+h] 16 KB
  __shared__ __bf16 k_lds[2][32][64];        // double-buffered K tile, 8 KB
  __shared__ __bf16 v_lds[2][64][32];        // double-buffered V tile, 8 KB

  const int b    = blockIdx.x >> 6;
  const int n0   = (blockIdx.x & 63) << 4;
  const int h    = threadIdx.x >> 5;          // wave == head
  const int lane = threadIdx.x & 31;
  const int lrow = lane & 15;
  const int halfA = (lane >> 4) << 4;
  const int halfB = (lane >> 4) << 4;
  const int mhi  = (lane >> 4) << 3;

  const __bf16* kbb = k_ws + (size_t)b * N_ * KD_;
  const __bf16* vbb = v_ws + (size_t)b * VD_ * N_;

  // Pull this wave's Wo rows toward the caches for the epilogue GEMM.
  __builtin_prefetch((const void*)(wo_bf + ((size_t)(h * 32) + lane) * (VD_ * NH_)), 0, 1);

  // Q B-operand for this head / n-tile (K = KD = 64 -> two ksteps), in regs.
  const __bf16* qbase = q_ws + (((size_t)(b * NH_ + h) * N_) + n0 + lrow) * KD_;
  const v16bf qb0 = loadB(qbase + 0  + halfB);
  const v16bf qb1 = loadB(qbase + 32 + halfB);

  // Cooperative async stage of one (K,V) tile pair: 2 x 4 KB, 2 loads/thread.
  auto stage_tile = [&](int buf, int m0) {
    const int t = (int)threadIdx.x;
    int kr = t >> 3, kc = (t & 7) * 8;        // K: 32 rows x 64 cols
    stage16(kbb + (size_t)(m0 + kr) * KD_ + kc, &k_lds[buf][kr][kc]);
    int vr = t >> 2, vc = (t & 3) * 8;        // V: 64 rows x 32 cols
    stage16(vbb + (size_t)vr * N_ + m0 + vc, &v_lds[buf][vr][vc]);
  };

  v8f oacc[4] = {v8f{}, v8f{}, v8f{}, v8f{}};   // o[64 x 16] for this head

  stage_tile(0, 0);
  for (int it = 0; it < 32; ++it) {
    const int buf = it & 1;
    if (it + 1 < 32) {
      stage_tile(buf ^ 1, (it + 1) * 32);     // overlap next DMA with compute
      async_wait_le2();                        // current tile's 2 loads done
    } else {
      async_wait_le0();
    }
    __syncthreads();                           // all waves' chunks landed

    // ---- logits[m 32][n 16] for this head: two 16x16 sub-tiles, K = 64 ----
#pragma unroll
    for (int t = 0; t < 2; ++t) {
      const __bf16* krow = &k_lds[buf][t * 16 + lrow][0];
      v8f acc = {};
      acc = wmma_bf16(loadA(krow + 0  + halfA), qb0, acc);
      acc = wmma_bf16(loadA(krow + 32 + halfA), qb1, acc);
#pragma unroll
      for (int j = 0; j < 8; ++j) {
        int ml = t * 16 + j + mhi;
        logits_lds[ml][lrow][h] = acc[j] * 0.125f;   // 1/sqrt(KD)
      }
    }
    __syncthreads();

    // ---- softmax over the 8 heads; 512 (m,n) positions / 256 threads ----
#pragma unroll
    for (int p = 0; p < 2; ++p) {
      int pos = (int)threadIdx.x + p * 256;
      int ml = pos >> 4, nl = pos & 15;
      const float* lg = &logits_lds[ml][nl][0];
      float mx = lg[0];
#pragma unroll
      for (int i = 1; i < 8; ++i) mx = fmaxf(mx, lg[i]);
      float e[8]; float s = 0.f;
#pragma unroll
      for (int i = 0; i < 8; ++i) { e[i] = __expf(lg[i] - mx); s += e[i]; }
      float rs = 1.0f / s;
#pragma unroll
      for (int i = 0; i < 8; ++i) attn_lds[i][nl][ml] = (__bf16)(e[i] * rs);
    }
    __syncthreads();

    // ---- o[v 64][n 16] += V[v][m 32] x attn[m 32][n 16], K = 32 over m ----
    const v16bf bm = loadB(&attn_lds[h][lrow][halfB]);
#pragma unroll
    for (int c = 0; c < 4; ++c) {
      oacc[c] = wmma_bf16(loadA(&v_lds[buf][c * 16 + lrow][halfA]), bm, oacc[c]);
    }
    __syncthreads();   // protect v_lds[buf]/attn_lds before next overwrite
  }

  // ---- stage O as bf16 [n 16][r=v*8+h 512] (matches Wo's native layout) ----
#pragma unroll
  for (int c = 0; c < 4; ++c) {
#pragma unroll
    for (int j = 0; j < 8; ++j) {
      int vrow = c * 16 + j + mhi;
      o_lds[lrow][vrow * 8 + h] = (__bf16)oacc[c][j];
    }
  }
  __syncthreads();

  // ---- projection: res[256 d][16 n] = Wo[256 x 512] x O[512 x 16] ----
  const int wave = h;
#pragma unroll
  for (int t = 0; t < 2; ++t) {
    const int d0 = wave * 32 + t * 16;
    const __bf16* worow = wo_bf + (size_t)(d0 + lrow) * (VD_ * NH_);
    const __bf16* obase = &o_lds[lrow][0];
    v8f acc = {};
#pragma unroll
    for (int ks = 0; ks < 16; ++ks) {        // K = 512 -> 16 steps of 32
      acc = wmma_bf16(loadA(worow + ks * 32 + halfA),
                      loadB(obase + ks * 32 + halfB), acc);
    }
#pragma unroll
    for (int j = 0; j < 8; ++j) {
      int d = d0 + j + mhi;
      size_t idx = ((size_t)b * C_ + d) * N_ + n0 + lrow;
      out[idx] = acc[j] * ls[d] + x[idx];    // LayerScale + residual, fp32
    }
  }
}

// ---------------------------------------------------------------------------
// Host launcher. Workspace layout (bytes):
//   xnT  @ 0         : 8*1024*256*2  = 4 MiB
//   q_ws @ 4194304   : 8*8*1024*64*2 = 8 MiB
//   k_ws @ 12582912  : 8*1024*64*2   = 1 MiB
//   v_ws @ 13631488  : 8*64*1024*2   = 1 MiB
//   wqT  @ 14680064  : 512*256*2
//   wkT  @ 14942208  : 64*256*2
//   wvT  @ 14974976  : 64*256*2
//   wo   @ 15007744  : 256*512*2      -> total ~14.6 MiB
// ---------------------------------------------------------------------------
extern "C" void kernel_launch(void* const* d_in, const int* in_sizes, int n_in,
                              void* d_out, int out_size, void* d_ws, size_t ws_size,
                              hipStream_t stream) {
  (void)in_sizes; (void)n_in; (void)out_size; (void)ws_size;
  const float* x   = (const float*)d_in[0];
  const float* g   = (const float*)d_in[1];
  const float* be  = (const float*)d_in[2];
  const float* mu  = (const float*)d_in[3];
  const float* var = (const float*)d_in[4];
  const float* Wq  = (const float*)d_in[5];
  const float* Wk  = (const float*)d_in[6];
  const float* Wv  = (const float*)d_in[7];
  const float* Wo  = (const float*)d_in[8];
  const float* ls  = (const float*)d_in[9];
  float* out = (float*)d_out;

  char* ws = (char*)d_ws;
  __bf16* xnT   = (__bf16*)(ws + 0);
  __bf16* q_ws  = (__bf16*)(ws + 4194304);
  __bf16* k_ws  = (__bf16*)(ws + 12582912);
  __bf16* v_ws  = (__bf16*)(ws + 13631488);
  __bf16* wqT   = (__bf16*)(ws + 14680064);
  __bf16* wkT   = (__bf16*)(ws + 14942208);
  __bf16* wvT   = (__bf16*)(ws + 14974976);
  __bf16* wo_bf = (__bf16*)(ws + 15007744);

  prep_xnT_kernel<<<(B_ * N_ * C_ + 255) / 256, 256, 0, stream>>>(x, g, be, mu, var, xnT);
  prep_w_kernel<<<1152, 256, 0, stream>>>(Wq, Wk, Wv, Wo, wqT, wkT, wvT, wo_bf);
  qkv_kernel<<<B_ * 64, 256, 0, stream>>>(xnT, wqT, wkT, wvT, q_ws, k_ws, v_ws);
  attn_kernel<<<B_ * 64, 256, 0, stream>>>(q_ws, k_ws, v_ws, wo_bf, x, ls, out);
}
